// MultiModalMaterialClassifier_31714038514073
// MI455X (gfx1250) — compile-verified, matrix operand order checked by reading, not run on hardware
//
#include <hip/hip_runtime.h>
#include <hip/hip_bf16.h>

// ---------------- problem constants (match reference) ----------------
#define NC   50000
#define ECRY 400000
#define FCRY 92
#define NKG  10000
#define EKG  100000
#define FKG  10
#define BSZ  32
#define HID  64
#define OUTD 128
#define LAY  8

typedef __attribute__((ext_vector_type(16))) _Float16 v16h;
typedef __attribute__((ext_vector_type(8)))  float    v8f;

#define FLAG_BIAS  1
#define FLAG_ACCUM 2
#define FLAG_RELU  4

// =====================================================================
// WMMA lane layouts per CDNA5 ISA 7.12.2 (f16 16x16x32):
//   A 16x32 : lanes 0-15 -> M=lane, lanes 16-31 -> M=lane-16
//             elem j<8  -> k = j   + 8*half ; elem j>=8 -> k = j+8 + 8*half
//   B 32x16 : same index mapping with N = lane&15
//   C/D     : vgpr r -> M = r + 8*half, N = lane&15
// =====================================================================

// Load one 16-element f16 A fragment from a full 32-wide K chunk.
// ap = &A[row, kk + 8*half]; elems 0..7 = ap[0..7], elems 8..15 = ap[16..23].
__device__ inline v16h load_a_chunk(const float* __restrict__ ap) {
  const float4* p = (const float4*)ap;
  float4 x0 = p[0], x1 = p[1], x2 = p[4], x3 = p[5];
  v16h a;
  a[0] = (_Float16)x0.x;  a[1] = (_Float16)x0.y;  a[2] = (_Float16)x0.z;  a[3] = (_Float16)x0.w;
  a[4] = (_Float16)x1.x;  a[5] = (_Float16)x1.y;  a[6] = (_Float16)x1.z;  a[7] = (_Float16)x1.w;
  a[8] = (_Float16)x2.x;  a[9] = (_Float16)x2.y;  a[10] = (_Float16)x2.z; a[11] = (_Float16)x2.w;
  a[12] = (_Float16)x3.x; a[13] = (_Float16)x3.y; a[14] = (_Float16)x3.z; a[15] = (_Float16)x3.w;
  return a;
}

// ---------------------------------------------------------------------
// Pack W[K,N] (f32, row-major) into WMMA-fragment-major f16:
//   Wp[(kc*nt16 + tt)*512 + lane*16 + j]  holds the B-fragment element j
//   of lane `lane` for the 32-K chunk kc and 16-column tile tt.
// Ragged K is zero-padded here, so the GEMM hot loop has no guards.
// ---------------------------------------------------------------------
__global__ void k_pack_w(const float* __restrict__ W, _Float16* __restrict__ Wp,
                         int K, int N, int nt16, int total) {
  int t = blockIdx.x * blockDim.x + threadIdx.x;
  if (t >= total) return;
  int j    = t & 15;
  int lane = (t >> 4) & 31;
  int frag = t >> 9;            // kc*nt16 + tt
  int tt   = frag % nt16;
  int kc   = frag / nt16;
  int half = lane >> 4, n = lane & 15;
  int k   = kc * 32 + 8 * half + (j < 8 ? j : j + 8);
  int col = tt * 16 + n;
  float v = (k < K) ? W[(size_t)k * N + col] : 0.0f;
  Wp[t] = (_Float16)v;
}

// ---------------------------------------------------------------------
// FAST GEMM: requires M%16==0, N%64==0. One wave -> 16 rows x 64 cols
// (4 WMMA accumulators). A: vectorized f32 loads + cvt_pk; B: one
// contiguous 32-byte v16h load per fragment from the packed buffer.
// ---------------------------------------------------------------------
__global__ void k_gemm_fast(const float* __restrict__ A, const _Float16* __restrict__ Wp,
                            const float* __restrict__ bias, float* __restrict__ C,
                            int M, int N, int K, int flags, int ngroups) {
  int wave = blockIdx.x * (blockDim.x >> 5) + (threadIdx.x >> 5);
  int total = (M >> 4) * ngroups;
  if (wave >= total) return;
  int tm = wave / ngroups;
  int tg = wave - tm * ngroups;
  int lane = threadIdx.x & 31;
  int half = lane >> 4;
  int l15  = lane & 15;
  const float* __restrict__ Arow = A + (size_t)(tm * 16 + l15) * K;
  int ncol0 = tg * 64 + l15;
  int nt16  = N >> 4;

  v8f acc[4] = {};
  if (flags & FLAG_ACCUM) {
#pragma unroll
    for (int r = 0; r < 8; ++r) {
      size_t row = (size_t)(tm * 16 + r + half * 8) * N + ncol0;
#pragma unroll
      for (int g = 0; g < 4; ++g) acc[g][r] = C[row + g * 16];
    }
  }

  // base of this wave's packed fragments (kc=0, tile tt=tg*4)
  const _Float16* __restrict__ wbase = Wp + (((size_t)tg * 4) << 9) + (lane << 4);
  size_t kcstride = (size_t)nt16 << 9;   // halves per 32-K chunk of fragments

  int nfull = K >> 5;
  for (int kc = 0; kc < nfull; ++kc) {
    v16h a = load_a_chunk(Arow + kc * 32 + half * 8);
    const _Float16* __restrict__ bp = wbase + (size_t)kc * kcstride;
#pragma unroll
    for (int g = 0; g < 4; ++g) {
      v16h b = *(const v16h*)(bp + ((size_t)g << 9));
      acc[g] = __builtin_amdgcn_wmma_f32_16x16x32_f16(false, a, false, b,
                                                      (short)0, acc[g], false, false);
    }
  }
  if ((nfull << 5) < K) {  // ragged-K tail: clamped A loads, packed B already padded
    int kk = nfull << 5;
    v16h a;
#pragma unroll
    for (int j = 0; j < 8; ++j) {
      int k0 = kk + half * 8 + j, k1 = k0 + 16;
      float v0 = Arow[k0 < K ? k0 : 0];
      float v1 = Arow[k1 < K ? k1 : 0];
      a[j]     = (_Float16)(k0 < K ? v0 : 0.0f);
      a[j + 8] = (_Float16)(k1 < K ? v1 : 0.0f);
    }
    const _Float16* __restrict__ bp = wbase + (size_t)nfull * kcstride;
#pragma unroll
    for (int g = 0; g < 4; ++g) {
      v16h b = *(const v16h*)(bp + ((size_t)g << 9));
      acc[g] = __builtin_amdgcn_wmma_f32_16x16x32_f16(false, a, false, b,
                                                      (short)0, acc[g], false, false);
    }
  }

#pragma unroll
  for (int r = 0; r < 8; ++r) {
    size_t row = (size_t)(tm * 16 + r + half * 8) * N + ncol0;
#pragma unroll
    for (int g = 0; g < 4; ++g) {
      float v = acc[g][r];
      if (flags & FLAG_BIAS) v += bias[ncol0 + g * 16];
      if (flags & FLAG_RELU) v = fmaxf(v, 0.0f);
      C[row + g * 16] = v;
    }
  }
}

// ---------------------------------------------------------------------
// Guarded GEMM (only for the ragged-N 32x3 classifier head)
// ---------------------------------------------------------------------
__global__ void k_gemm_guard(const float* __restrict__ A, const float* __restrict__ W,
                             const float* __restrict__ bias, float* __restrict__ C,
                             int M, int N, int K, int flags, int tiles_n) {
  int wave = blockIdx.x * (blockDim.x >> 5) + (threadIdx.x >> 5);
  int tiles_m = (M + 15) >> 4;
  if (wave >= tiles_m * tiles_n) return;
  int tm = wave / tiles_n;
  int tn = wave - tm * tiles_n;
  int lane = threadIdx.x & 31;
  int half = lane >> 4;
  int l15  = lane & 15;
  int mr = (tm << 4) + l15;
  int nc = (tn << 4) + l15;
  bool mok = mr < M, nok = nc < N;

  v8f acc = {};
  if (flags & FLAG_ACCUM) {
#pragma unroll
    for (int r = 0; r < 8; ++r) {
      int row = (tm << 4) + r + (half << 3);
      if (row < M && nok) acc[r] = C[(size_t)row * N + nc];
    }
  }
  for (int kk = 0; kk < K; kk += 32) {
    v16h a, b;
#pragma unroll
    for (int j = 0; j < 8; ++j) {
      int k0 = kk + (half << 3) + j, k1 = k0 + 16;
      int k0c = k0 < K ? k0 : 0, k1c = k1 < K ? k1 : 0;
      float a0 = mok ? A[(size_t)mr * K + k0c] : 0.0f;
      float a1 = mok ? A[(size_t)mr * K + k1c] : 0.0f;
      a[j]     = (_Float16)((mok && k0 < K) ? a0 : 0.0f);
      a[j + 8] = (_Float16)((mok && k1 < K) ? a1 : 0.0f);
      float b0 = nok ? W[(size_t)k0c * N + nc] : 0.0f;
      float b1 = nok ? W[(size_t)k1c * N + nc] : 0.0f;
      b[j]     = (_Float16)((nok && k0 < K) ? b0 : 0.0f);
      b[j + 8] = (_Float16)((nok && k1 < K) ? b1 : 0.0f);
    }
    acc = __builtin_amdgcn_wmma_f32_16x16x32_f16(false, a, false, b,
                                                 (short)0, acc, false, false);
  }
#pragma unroll
  for (int r = 0; r < 8; ++r) {
    int row = (tm << 4) + r + (half << 3);
    if (row < M && nok) {
      float v = acc[r];
      if (flags & FLAG_BIAS) v += bias[nc];
      if (flags & FLAG_RELU) v = fmaxf(v, 0.0f);
      C[(size_t)row * N + nc] = v;
    }
  }
}

// ---------------- elementwise / reduction helpers ----------------
__global__ void k_fill(float* p, int n, float v) {
  int t = blockIdx.x * blockDim.x + threadIdx.x;
  if (t < n) p[t] = v;
}

__global__ void k_ln_relu(const float* __restrict__ x, const float* __restrict__ g,
                          const float* __restrict__ b, float* __restrict__ y,
                          int rows, int D) {
  int row = blockIdx.x;
  if (row >= rows) return;
  const float* xr = x + (size_t)row * D;
  float s = 0.0f, sq = 0.0f;
  for (int i = threadIdx.x; i < D; i += blockDim.x) {
    float v = xr[i]; s += v; sq += v * v;
  }
  __shared__ float sh[128];
  sh[threadIdx.x] = s; __syncthreads();
  for (int off = 64; off; off >>= 1) {
    if (threadIdx.x < off) sh[threadIdx.x] += sh[threadIdx.x + off];
    __syncthreads();
  }
  float mean = sh[0] / (float)D;
  __syncthreads();
  sh[threadIdx.x] = sq; __syncthreads();
  for (int off = 64; off; off >>= 1) {
    if (threadIdx.x < off) sh[threadIdx.x] += sh[threadIdx.x + off];
    __syncthreads();
  }
  float var = sh[0] / (float)D - mean * mean;
  float rstd = rsqrtf(var + 1e-5f);
  for (int i = threadIdx.x; i < D; i += blockDim.x) {
    float v = (xr[i] - mean) * rstd * g[i] + b[i];
    y[(size_t)row * D + i] = fmaxf(v, 0.0f);
  }
}

// ---------------- degree precompute (crystal graph) ----------------
__global__ void k_indeg(const int* __restrict__ dst, int nE, float* cnt) {
  int t = blockIdx.x * blockDim.x + threadIdx.x;
  if (t < nE) atomicAdd(&cnt[dst[t]], 1.0f);
}
__global__ void k_deg_fin(float* cnt, float* di, int n) {
  int t = blockIdx.x * blockDim.x + threadIdx.x;
  if (t >= n) return;
  float c = cnt[t];
  di[t]  = rsqrtf(c + 1.0f);          // GCN: rsqrt(indeg + self-loop)
  cnt[t] = 1.0f / fmaxf(c, 1.0f);     // SAGE: 1/max(indeg,1)
}

// ---------------- GCN ----------------
__global__ void k_gcn_init(const float* __restrict__ h, const float* __restrict__ di,
                           const float* __restrict__ bias, float* out, int n) {
  int t = blockIdx.x * blockDim.x + threadIdx.x;
  if (t >= n * HID) return;
  int node = t >> 6, f = t & 63;
  float d = di[node];
  out[t] = h[t] * d * d + bias[f];
}
__global__ void k_gcn_edge(const int* __restrict__ src, const int* __restrict__ dst, int nE,
                           const float* __restrict__ h, const float* __restrict__ di,
                           float* out) {
  int t = blockIdx.x * blockDim.x + threadIdx.x;
  if (t >= nE * HID) return;
  int e = t >> 6, f = t & 63;
  int s = src[e], d = dst[e];
  atomicAdd(&out[(size_t)d * HID + f], h[(size_t)s * HID + f] * di[s] * di[d]);
}

// ---------------- segment-softmax machinery (GAT + Transformer) ----------------
__device__ inline unsigned fenc(float f) {
  unsigned bts = __float_as_uint(f);
  return (bts & 0x80000000u) ? ~bts : (bts | 0x80000000u);
}
__device__ inline float fdec(unsigned k) {
  unsigned bts = (k & 0x80000000u) ? (k & 0x7FFFFFFFu) : ~k;
  return __uint_as_float(bts);
}

__global__ void k_gat_node_attn(const float* __restrict__ h, const float* __restrict__ pas,
                                const float* __restrict__ pad, float* out_s, float* out_d,
                                int n) {
  int t = blockIdx.x * blockDim.x + threadIdx.x;
  if (t >= n * 8) return;
  int node = t >> 3, hd = t & 7;
  const float* hp = h + (size_t)node * HID + hd * 8;
  float ss = 0.0f, sd = 0.0f;
#pragma unroll
  for (int i = 0; i < 8; ++i) { ss += hp[i] * pas[hd * 8 + i]; sd += hp[i] * pad[hd * 8 + i]; }
  out_s[t] = ss; out_d[t] = sd;
}

__global__ void k_gat_logit(const int* __restrict__ src, const int* __restrict__ dst,
                            int nE, int nSelf, const float* __restrict__ as_,
                            const float* __restrict__ ad_, float* eatt, unsigned* smx) {
  int t = blockIdx.x * blockDim.x + threadIdx.x;
  if (t >= (nE + nSelf) * 8) return;
  int e = t >> 3, hd = t & 7;
  int s = (e < nE) ? src[e] : (e - nE);
  int d = (e < nE) ? dst[e] : (e - nE);
  float x = as_[s * 8 + hd] + ad_[d * 8 + hd];
  float lg = (x > 0.0f) ? x : 0.2f * x;
  eatt[t] = lg;
  atomicMax(&smx[d * 8 + hd], fenc(lg));
}

__global__ void k_tr_logit(const int* __restrict__ src, const int* __restrict__ dst, int nE,
                           const float* __restrict__ q, const float* __restrict__ kbuf,
                           float* eatt, unsigned* smx) {
  int t = blockIdx.x * blockDim.x + threadIdx.x;
  if (t >= nE * 8) return;
  int e = t >> 3, hd = t & 7;
  int s = src[e], d = dst[e];
  const float* qp = q + (size_t)d * HID + hd * 8;
  const float* kp = kbuf + (size_t)s * HID + hd * 8;
  float dot = 0.0f;
#pragma unroll
  for (int i = 0; i < 8; ++i) dot += qp[i] * kp[i];
  float lg = dot * 0.35355339059327373f;
  eatt[t] = lg;
  atomicMax(&smx[d * 8 + hd], fenc(lg));
}

__global__ void k_attn_exp(const int* __restrict__ dst, int nE, int nSelf,
                           float* eatt, const unsigned* __restrict__ smx, float* ssum) {
  int t = blockIdx.x * blockDim.x + threadIdx.x;
  if (t >= (nE + nSelf) * 8) return;
  int e = t >> 3, hd = t & 7;
  int d = (e < nE) ? dst[e] : (e - nE);
  float m = fdec(smx[d * 8 + hd]);
  if ((__float_as_uint(m) & 0x7F800000u) == 0x7F800000u) m = 0.0f;
  float ev = __expf(eatt[t] - m);
  eatt[t] = ev;
  atomicAdd(&ssum[d * 8 + hd], ev);
}

__global__ void k_attn_scatter(const int* __restrict__ src, const int* __restrict__ dst,
                               int nE, int nSelf, const float* __restrict__ eatt,
                               const float* __restrict__ ssum, const float* __restrict__ val,
                               float* out) {
  int t = blockIdx.x * blockDim.x + threadIdx.x;
  if (t >= (nE + nSelf) * HID) return;
  int e = t >> 6, f = t & 63, hd = f >> 3;
  int s = (e < nE) ? src[e] : (e - nE);
  int d = (e < nE) ? dst[e] : (e - nE);
  float alpha = eatt[e * 8 + hd] / (ssum[d * 8 + hd] + 1e-16f);
  atomicAdd(&out[(size_t)d * HID + f], alpha * val[(size_t)s * HID + f]);
}

__global__ void k_rowbias(float* out, const float* __restrict__ bias, int n) {
  int t = blockIdx.x * blockDim.x + threadIdx.x;
  if (t < n * HID) out[t] = bias[t & 63];
}

// ---------------- SAGE ----------------
__global__ void k_sage_scatter(const int* __restrict__ src, const int* __restrict__ dst, int nE,
                               const float* __restrict__ x, float* agg) {
  int t = blockIdx.x * blockDim.x + threadIdx.x;
  if (t >= nE * HID) return;
  int e = t >> 6, f = t & 63;
  atomicAdd(&agg[(size_t)dst[e] * HID + f], x[(size_t)src[e] * HID + f]);
}
__global__ void k_scale_rows(float* x, const float* __restrict__ sc, int n) {
  int t = blockIdx.x * blockDim.x + threadIdx.x;
  if (t < n * HID) x[t] *= sc[t >> 6];
}

// ---------------- Transformer beta-gate ----------------
__global__ void k_tr_gate(const float* __restrict__ msg, const float* __restrict__ r,
                          const float* __restrict__ Wb, const float* __restrict__ bb,
                          float* out, int n) {
  int node = blockIdx.x * blockDim.x + threadIdx.x;
  if (node >= n) return;
  const float* mp = msg + (size_t)node * HID;
  const float* rp = r + (size_t)node * HID;
  float acc = bb[0];
  for (int i = 0; i < HID; ++i) {
    float m = mp[i], rr = rp[i];
    acc += m * Wb[i] + rr * Wb[HID + i] + (m - rr) * Wb[2 * HID + i];
  }
  float g = 1.0f / (1.0f + __expf(-acc));
  for (int i = 0; i < HID; ++i)
    out[(size_t)node * HID + i] = g * rp[i] + (1.0f - g) * mp[i];
}

// ---------------- mean pooling ----------------
__global__ void k_pool_scatter(const float* __restrict__ x, const int* __restrict__ batch,
                               int n, float* pooled) {
  int t = blockIdx.x * blockDim.x + threadIdx.x;
  if (t >= n * HID) return;
  int node = t >> 6, f = t & 63;
  atomicAdd(&pooled[(size_t)batch[node] * HID + f], x[t]);
}
__global__ void k_pool_count(const int* __restrict__ batch, int n, float* cnt) {
  int t = blockIdx.x * blockDim.x + threadIdx.x;
  if (t < n) atomicAdd(&cnt[batch[t]], 1.0f);
}
__global__ void k_pool_div(float* pooled, const float* __restrict__ cnt) {
  int t = blockIdx.x * blockDim.x + threadIdx.x;
  if (t < BSZ * HID) pooled[t] /= fmaxf(cnt[t >> 6], 1.0f);
}

// ---------------- fusion gate + concat ----------------
__global__ void k_fuse(const float* __restrict__ feats, const float* __restrict__ gate,
                       float* fused) {
  int t = blockIdx.x * blockDim.x + threadIdx.x;
  if (t >= BSZ * 6 * OUTD) return;
  int bi = t / (6 * OUTD), c = t - bi * 6 * OUTD;
  int i = c >> 7, j = c & 127;
  float g = 1.0f / (1.0f + __expf(-gate[i]));
  fused[t] = feats[(size_t)i * BSZ * OUTD + (size_t)bi * OUTD + j] * g;
}

// =====================================================================
// Host orchestration.
// Input index map (setup_inputs insertion order; params pytree flattened
// with jax's sorted-dict-key order, ASCII => capitals first):
//   0 x_crys  1 edge_crys  2 batch_crys  3 x_k  4 edge_k  5 batch_k
//   6 scalar_feats  7 decomp_feats
//   8 C1b  9 C1w  10 C2b  11 C2w
//   12..19 decomp: W1 W2 b1 b2 be1 be2 g1 g2
//   20..29 gat:    Fb Fw Pb Pw W a_d a_s b lnb lng
//   30 gate
//   31..38 gcn:    Fb Fw Pb Pw W b lnb lng
//   39..47 sage:   Fb Fw Pb Pw Wl Wr b lnb lng
//   48..55 scalar: W1 W2 b1 b2 be1 be2 g1 g2
//   56..71 tr:     Fb Fw Pb Pw Wb Wk Wq Ws Wv bb bk bq bs bv lnb lng
// =====================================================================
extern "C" void kernel_launch(void* const* d_in, const int* in_sizes, int n_in,
                              void* d_out, int out_size, void* d_ws, size_t ws_size,
                              hipStream_t stream) {
  (void)in_sizes; (void)n_in; (void)out_size; (void)ws_size;
  auto F = [&](int i) { return (const float*)d_in[i]; };
  auto I = [&](int i) { return (const int*)d_in[i]; };

  const float* x_crys = F(0);
  const int*   e_crys = I(1);
  const int*   b_crys = I(2);
  const float* x_k    = F(3);
  const int*   e_k    = I(4);
  const int*   b_k    = I(5);
  const int* srcC = e_crys;          const int* dstC = e_crys + ECRY;
  const int* srcK = e_k;             const int* dstK = e_k + EKG;

  // ---- workspace carve-out (floats; every size is a multiple of 16) ----
  float* ws = (float*)d_ws;
  size_t off = 0;
  auto carve = [&](size_t n) { float* p = ws + off; off += n; return p; };
  float* xa   = carve((size_t)NC * HID);
  float* xb   = carve((size_t)NC * HID);
  float* xh   = carve((size_t)NC * HID);
  float* di   = carve(NC);
  float* sdi  = carve(NC);
  float* as_  = carve((size_t)NC * 8);
  float* ad_  = carve((size_t)NC * 8);
  float* smxf = carve((size_t)NC * 8);
  float* ssum = carve((size_t)NC * 8);
  float* eatt = carve((size_t)(ECRY + NC) * 8);
  float* qb   = carve((size_t)NKG * HID);
  float* kb   = carve((size_t)NKG * HID);
  float* vb   = carve((size_t)NKG * HID);
  float* rb   = carve((size_t)NKG * HID);
  float* msg  = carve((size_t)NKG * HID);
  float* pooled = carve((size_t)BSZ * HID);
  float* pcnt   = carve(BSZ);
  float* feats  = carve((size_t)6 * BSZ * OUTD);
  float* fused  = carve((size_t)BSZ * 6 * OUTD);
  float* h1     = carve((size_t)BSZ * 256);
  float* fe_tmp = carve((size_t)BSZ * 256);
  float* wpackf = carve(98304);      // 196608 f16 = max packed weight (768x256)
  unsigned*  smx   = (unsigned*)smxf;
  _Float16*  wpack = (_Float16*)wpackf;

  auto gb = [](int n) { return (n + 255) / 256; };
  auto gemm = [&](const float* A, const float* W, const float* bias, float* C,
                  int M, int N, int K, int flags) {
    if ((M % 16 == 0) && (N % 64 == 0)) {
      int nchunks = (K + 31) >> 5;
      int nt16 = N >> 4;
      int tot = nchunks * nt16 * 512;
      k_pack_w<<<gb(tot), 256, 0, stream>>>(W, wpack, K, N, nt16, tot);
      int ngroups = N >> 6;
      int waves = (M >> 4) * ngroups;
      k_gemm_fast<<<(waves + 7) / 8, 256, 0, stream>>>(A, wpack, bias, C, M, N, K,
                                                       flags, ngroups);
    } else {
      int tn = (N + 15) >> 4;
      int waves = ((M + 15) >> 4) * tn;
      k_gemm_guard<<<(waves + 3) / 4, 128, 0, stream>>>(A, W, bias, C, M, N, K, flags, tn);
    }
  };
  auto fill0 = [&](float* p, int n) { k_fill<<<gb(n), 256, 0, stream>>>(p, n, 0.0f); };
  auto lnrelu = [&](const float* x, const float* g, const float* b, float* y, int rows, int D) {
    k_ln_relu<<<rows, 128, 0, stream>>>(x, g, b, y, rows, D);
  };
  auto pool = [&](const float* x, const int* batch, int n, const float* Fw,
                  const float* Fb, float* slot) {
    fill0(pooled, BSZ * HID);
    fill0(pcnt, BSZ);
    k_pool_scatter<<<gb(n * HID), 256, 0, stream>>>(x, batch, n, pooled);
    k_pool_count<<<gb(n), 256, 0, stream>>>(batch, n, pcnt);
    k_pool_div<<<gb(BSZ * HID), 256, 0, stream>>>(pooled, pcnt);
    gemm(pooled, Fw, Fb, slot, BSZ, OUTD, HID, FLAG_BIAS);
  };

  // ---- crystal-graph degrees (shared by GCN & SAGE) ----
  fill0(sdi, NC);
  k_indeg<<<gb(ECRY), 256, 0, stream>>>(dstC, ECRY, sdi);
  k_deg_fin<<<gb(NC), 256, 0, stream>>>(sdi, di, NC);

  // =================== GCN encoder (feats slot 0) ===================
  gemm(x_crys, F(34), F(33), xa, NC, HID, FCRY, FLAG_BIAS);
  for (int l = 0; l < LAY; ++l) {
    gemm(xa, F(35) + (size_t)l * HID * HID, nullptr, xh, NC, HID, HID, 0);
    k_gcn_init<<<gb(NC * HID), 256, 0, stream>>>(xh, di, F(36) + l * HID, xb, NC);
    k_gcn_edge<<<gb(ECRY * HID), 256, 0, stream>>>(srcC, dstC, ECRY, xh, di, xb);
    lnrelu(xb, F(38) + l * HID, F(37) + l * HID, xa, NC, HID);
  }
  pool(xa, b_crys, NC, F(32), F(31), feats + 0 * BSZ * OUTD);

  // =================== GAT encoder (feats slot 1) ===================
  gemm(x_crys, F(23), F(22), xa, NC, HID, FCRY, FLAG_BIAS);
  for (int l = 0; l < LAY; ++l) {
    gemm(xa, F(24) + (size_t)l * HID * HID, nullptr, xh, NC, HID, HID, 0);
    k_gat_node_attn<<<gb(NC * 8), 256, 0, stream>>>(xh, F(26) + l * 64, F(25) + l * 64,
                                                    as_, ad_, NC);
    fill0(smxf, NC * 8);
    fill0(ssum, NC * 8);
    k_gat_logit<<<gb((ECRY + NC) * 8), 256, 0, stream>>>(srcC, dstC, ECRY, NC, as_, ad_,
                                                         eatt, smx);
    k_attn_exp<<<gb((ECRY + NC) * 8), 256, 0, stream>>>(dstC, ECRY, NC, eatt, smx, ssum);
    k_rowbias<<<gb(NC * HID), 256, 0, stream>>>(xb, F(27) + l * HID, NC);
    k_attn_scatter<<<gb((ECRY + NC) * HID), 256, 0, stream>>>(srcC, dstC, ECRY, NC, eatt,
                                                              ssum, xh, xb);
    lnrelu(xb, F(29) + l * HID, F(28) + l * HID, xa, NC, HID);
  }
  pool(xa, b_crys, NC, F(21), F(20), feats + 1 * BSZ * OUTD);

  // =================== SAGE encoder (feats slot 2) ===================
  gemm(x_crys, F(42), F(41), xa, NC, HID, FCRY, FLAG_BIAS);
  for (int l = 0; l < LAY; ++l) {
    fill0(xh, NC * HID);
    k_sage_scatter<<<gb(ECRY * HID), 256, 0, stream>>>(srcC, dstC, ECRY, xa, xh);
    k_scale_rows<<<gb(NC * HID), 256, 0, stream>>>(xh, sdi, NC);
    gemm(xh, F(43) + (size_t)l * HID * HID, nullptr, xb, NC, HID, HID, 0);
    gemm(xa, F(44) + (size_t)l * HID * HID, F(45) + l * HID, xb, NC, HID, HID,
         FLAG_BIAS | FLAG_ACCUM);
    lnrelu(xb, F(47) + l * HID, F(46) + l * HID, xa, NC, HID);
  }
  pool(xa, b_crys, NC, F(40), F(39), feats + 2 * BSZ * OUTD);

  // ============ TransformerConv encoder, k-graph (feats slot 3) ============
  gemm(x_k, F(59), F(58), xa, NKG, HID, FKG, FLAG_BIAS);
  for (int l = 0; l < LAY; ++l) {
    size_t wo = (size_t)l * HID * HID;
    gemm(xa, F(62) + wo, F(67) + l * HID, qb, NKG, HID, HID, FLAG_BIAS);  // Wq,bq
    gemm(xa, F(61) + wo, F(66) + l * HID, kb, NKG, HID, HID, FLAG_BIAS);  // Wk,bk
    gemm(xa, F(64) + wo, F(69) + l * HID, vb, NKG, HID, HID, FLAG_BIAS);  // Wv,bv
    gemm(xa, F(63) + wo, F(68) + l * HID, rb, NKG, HID, HID, FLAG_BIAS);  // Ws,bs
    fill0(smxf, NKG * 8);
    fill0(ssum, NKG * 8);
    k_tr_logit<<<gb(EKG * 8), 256, 0, stream>>>(srcK, dstK, EKG, qb, kb, eatt, smx);
    k_attn_exp<<<gb(EKG * 8), 256, 0, stream>>>(dstK, EKG, 0, eatt, smx, ssum);
    fill0(msg, NKG * HID);
    k_attn_scatter<<<gb(EKG * HID), 256, 0, stream>>>(srcK, dstK, EKG, 0, eatt, ssum, vb, msg);
    k_tr_gate<<<gb(NKG), 256, 0, stream>>>(msg, rb, F(60) + (size_t)l * 3 * HID, F(65) + l,
                                           xb, NKG);
    lnrelu(xb, F(71) + l * HID, F(70) + l * HID, xa, NKG, HID);
  }
  pool(xa, b_k, NKG, F(57), F(56), feats + 3 * BSZ * OUTD);

  // =================== feature encoders (slots 4, 5) ===================
  gemm(F(6), F(48), F(50), fe_tmp, BSZ, 256, 200, FLAG_BIAS);
  lnrelu(fe_tmp, F(54), F(52), fe_tmp, BSZ, 256);
  gemm(fe_tmp, F(49), F(51), feats + 4 * BSZ * OUTD, BSZ, OUTD, 256, FLAG_BIAS);
  lnrelu(feats + 4 * BSZ * OUTD, F(55), F(53), feats + 4 * BSZ * OUTD, BSZ, OUTD);
  gemm(F(7), F(12), F(14), fe_tmp, BSZ, 256, 100, FLAG_BIAS);
  lnrelu(fe_tmp, F(18), F(16), fe_tmp, BSZ, 256);
  gemm(fe_tmp, F(13), F(15), feats + 5 * BSZ * OUTD, BSZ, OUTD, 256, FLAG_BIAS);
  lnrelu(feats + 5 * BSZ * OUTD, F(19), F(17), feats + 5 * BSZ * OUTD, BSZ, OUTD);

  // =================== fusion classifier ===================
  k_fuse<<<gb(BSZ * 6 * OUTD), 256, 0, stream>>>(feats, F(30), fused);
  gemm(fused, F(9), F(8), h1, BSZ, 256, 6 * OUTD, FLAG_BIAS | FLAG_RELU);
  gemm(h1, F(11), F(10), (float*)d_out, BSZ, 3, 256, FLAG_BIAS);
}
